// InputMasker_28896539968071
// MI455X (gfx1250) — matrix operand order repaired and checked by reading, not compiled, not run on hardware
//
#include <hip/hip_runtime.h>
#include <stdint.h>

// ---------------------------------------------------------------------------
// Self-attention block (B=4, S=4096, D=768) for gfx1250 (MI455X).
//   K1: QKV projection, fp32 -> bf16 WMMA GEMM, V stored transposed.
//   K2: flash attention (online softmax) + LayerNorm + w_out dot, fused.
// Workspace layout (bf16 elements):
//   Qb  [16384][768]        @ 0
//   Kb  [16384][768]        @ 12,582,912
//   Vtb [4][768][4096]      @ 25,165,824      (total 75.5 MB)
// Both kernels software-pipeline operand fetch (double-buffered, fenced with
// sched_barrier) so global loads overlap the WMMA stream.
// ---------------------------------------------------------------------------

typedef __attribute__((ext_vector_type(16))) __bf16 v16bf;
typedef __attribute__((ext_vector_type(8)))  __bf16 v8bf;
typedef __attribute__((ext_vector_type(8)))  float  v8f;

#define LOG2E 1.44269504088896340736f

__device__ __forceinline__ v16bf cat8(v8bf a, v8bf b) {
  return __builtin_shufflevector(a, b, 0,1,2,3,4,5,6,7,8,9,10,11,12,13,14,15);
}
__device__ __forceinline__ v16bf cvt2bf16(v8f a, v8f b) {
  return cat8(__builtin_convertvector(a, v8bf), __builtin_convertvector(b, v8bf));
}
__device__ __forceinline__ v8f wmma_bf16(v16bf a, v16bf b, v8f c) {
  // (neg_a, A, neg_b, B, c_mod, C, reuse_a, reuse_b)
  return __builtin_amdgcn_wmma_f32_16x16x32_bf16(false, a, false, b, (short)0, c,
                                                 false, false);
}
// 16-lane (half-wave) butterfly reductions; xor masks 1,2,4,8 stay inside each
// 16-lane half, matching the C-tile row layout (rows m<8 in lanes 0-15, m>=8
// in lanes 16-31). Result replicated across the 16 lanes of each half.
__device__ __forceinline__ float redmax16(float v) {
  v = fmaxf(v, __shfl_xor(v, 1, 32));
  v = fmaxf(v, __shfl_xor(v, 2, 32));
  v = fmaxf(v, __shfl_xor(v, 4, 32));
  v = fmaxf(v, __shfl_xor(v, 8, 32));
  return v;
}
__device__ __forceinline__ float redsum16(float v) {
  v += __shfl_xor(v, 1, 32);
  v += __shfl_xor(v, 2, 32);
  v += __shfl_xor(v, 4, 32);
  v += __shfl_xor(v, 8, 32);
  return v;
}

// ---------------------------------------------------------------------------
// Kernel 1: Q/K/V = x @ W^T + b, output bf16. One wave: 16x64 tile of one of
// {Q,K,V}. Double-buffered operand fetch: step d0+32 is loaded while step d0
// runs cvt+WMMA, fenced by sched_barrier so the loads stay batched.
// ---------------------------------------------------------------------------
struct QkvOps { v8f a0, a1, b0[4], b1[4]; };  // 20 x b128 per k-step

__device__ __forceinline__ QkvOps qkv_load(const float* xrow, const float* W,
                                           int c0, int ln, int lh, int d0) {
  QkvOps o;
  o.a0 = *(const v8f*)(xrow + d0 + lh * 16);
  o.a1 = *(const v8f*)(xrow + d0 + 8 + lh * 16);
#pragma unroll
  for (int j = 0; j < 4; ++j) {
    const float* wr = W + (size_t)(c0 + j * 16 + ln) * 768 + d0 + lh * 16;
    o.b0[j] = *(const v8f*)wr;
    o.b1[j] = *(const v8f*)(wr + 8);
  }
  return o;
}
__device__ __forceinline__ void qkv_mma(const QkvOps& o, v8f acc[4]) {
  v16bf A = cvt2bf16(o.a0, o.a1);
#pragma unroll
  for (int j = 0; j < 4; ++j)
    acc[j] = wmma_bf16(A, cvt2bf16(o.b0[j], o.b1[j]), acc[j]);
}

__global__ __launch_bounds__(128) void qkv_proj_kernel(
    const float* __restrict__ x,
    const float* __restrict__ Wq, const float* __restrict__ bq,
    const float* __restrict__ Wk, const float* __restrict__ bk,
    const float* __restrict__ Wv, const float* __restrict__ bv,
    uint16_t* __restrict__ ws_u16)
{
  const int lane = threadIdx.x & 31;
  const int ln = lane & 15, lh = lane >> 4;
  const int wid = blockIdx.x * 4 + (threadIdx.x >> 5);
  const int z   = wid / (1024 * 12);           // 0=Q, 1=K, 2=V
  const int rem = wid % (1024 * 12);
  const int r0  = (rem / 12) * 16;             // row tile
  const int c0  = (rem % 12) * 64;             // column group

  const float* W    = (z == 0) ? Wq : (z == 1) ? Wk : Wv;
  const float* bias = (z == 0) ? bq : (z == 1) ? bk : bv;

  __bf16* outQ  = (__bf16*)ws_u16;                        // [16384][768]
  __bf16* outK  = outQ + (size_t)16384 * 768;             // [16384][768]
  __bf16* outVt = outK + (size_t)16384 * 768;             // [4][768][4096]

  v8f acc[4];
#pragma unroll
  for (int j = 0; j < 4; ++j)
#pragma unroll
    for (int g = 0; g < 8; ++g) acc[j][g] = 0.f;

  const float* xrow = x + (size_t)(r0 + ln) * 768;

  // Pipeline: 2 k-steps per iteration, operands fetched one step ahead.
  QkvOps cur = qkv_load(xrow, W, c0, ln, lh, 0);
  for (int d0 = 0; d0 < 768; d0 += 64) {
    QkvOps nxt = qkv_load(xrow, W, c0, ln, lh, d0 + 32);
    __builtin_amdgcn_sched_barrier(0);
    qkv_mma(cur, acc);
    const int d2 = (d0 + 64 < 768) ? d0 + 64 : 0;   // wrap: harmless prefetch
    cur = qkv_load(xrow, W, c0, ln, lh, d2);
    __builtin_amdgcn_sched_barrier(0);
    qkv_mma(nxt, acc);
  }

  const int bb = r0 >> 12;       // batch
  const int s0 = r0 & 4095;      // seq pos within batch
#pragma unroll
  for (int j = 0; j < 4; ++j) {
    const int c = c0 + j * 16 + ln;
    const float bc = bias[c];
#pragma unroll
    for (int g = 0; g < 8; ++g) {
      const int m = g + 8 * lh;  // C layout: lanes 0-15 -> rows g, 16-31 -> g+8
      const __bf16 h = (__bf16)(acc[j][g] + bc);
      if (z == 2)      outVt[((size_t)bb * 768 + c) * 4096 + s0 + m] = h;
      else if (z == 1) outK[(size_t)(r0 + m) * 768 + c] = h;
      else             outQ[(size_t)(r0 + m) * 768 + c] = h;
    }
  }
}

// ---------------------------------------------------------------------------
// Kernel 2: flash attention + LayerNorm + w_out dot.
// One wave owns 16 query rows: Q resident in 24 A-tiles (192 VGPRs),
// fp32 accumulator = 48 C-tiles (384 VGPRs). Per 32-key chunk:
//   48 WMMA (Q.K^T) -> online softmax -> P via 1KB LDS relayout -> 48 WMMA (P.V)
// K/V groups are double-buffered (next group's loads in flight during the
// current WMMA burst); acc rescale is interleaved into the P.V groups.
// ---------------------------------------------------------------------------
struct KGroup { v16bf a[8]; };   // 4 k-steps x {keys 0-15, keys 16-31}
struct VGroup { v16bf v[4]; };   // 4 d-tiles

__device__ __forceinline__ KGroup load_kgroup(const __bf16* K, int kc, int t0,
                                              int ln, int lh) {
  KGroup g;
#pragma unroll
  for (int u = 0; u < 4; ++u) {
    g.a[u]     = *(const v16bf*)(K + (size_t)(kc + ln) * 768 + (t0 + u) * 32 + lh * 16);
    g.a[4 + u] = *(const v16bf*)(K + (size_t)(kc + 16 + ln) * 768 + (t0 + u) * 32 + lh * 16);
  }
  return g;
}
__device__ __forceinline__ VGroup load_vgroup(const __bf16* Vt, int kc, int t0,
                                              int ln, int lh) {
  VGroup g;
#pragma unroll
  for (int u = 0; u < 4; ++u)
    g.v[u] = *(const v16bf*)(Vt + (size_t)((t0 + u) * 16 + ln) * 4096 + kc + lh * 16);
  return g;
}

__global__ __launch_bounds__(128) void flash_attn_ln_kernel(
    const uint16_t* __restrict__ ws_u16,
    const float* __restrict__ gamma, const float* __restrict__ beta,
    const float* __restrict__ w_out, const float* __restrict__ b_out,
    float* __restrict__ out)
{
  __shared__ __align__(64) __bf16 pbuf[4][16][32];  // per-wave P tile [m][k]
  const int lane = threadIdx.x & 31;
  const int wave = threadIdx.x >> 5;
  const int ln = lane & 15, lh = lane >> 4;
  const int wid = blockIdx.x * 4 + wave;
  const int row0 = wid * 16;                 // global query row block
  const int bb = row0 >> 12;                 // batch (no straddle)

  const __bf16* Qb  = (const __bf16*)ws_u16;
  const __bf16* Kb  = Qb + (size_t)16384 * 768;
  const __bf16* Vtb = Kb + (size_t)16384 * 768;
  const __bf16* Q   = Qb + (size_t)row0 * 768;
  const __bf16* K   = Kb + (size_t)bb * 4096 * 768;
  const __bf16* Vt  = Vtb + (size_t)bb * 768 * 4096;

  // Q tiles in registers: A-layout, lane ln = row, runs at d = t*32 + lh*16 (+8)
  v16bf q[24];
#pragma unroll
  for (int t = 0; t < 24; ++t) {
    const __bf16* qa = Q + (size_t)ln * 768 + t * 32 + lh * 16;
    q[t] = cat8(*(const v8bf*)qa, *(const v8bf*)(qa + 8));
  }

  v8f acc[48];
#pragma unroll
  for (int t = 0; t < 48; ++t)
#pragma unroll
    for (int g = 0; g < 8; ++g) acc[t][g] = 0.f;

  float ms[8], ls[8];
#pragma unroll
  for (int g = 0; g < 8; ++g) { ms[g] = -__builtin_inff(); ls[g] = 0.f; }

  for (int kc = 0; kc < 4096; kc += 32) {
    // ---- S = Q . K^T for 32 keys (two 16x16 C tiles), pipelined groups ----
    v8f s0, s1;
#pragma unroll
    for (int g = 0; g < 8; ++g) { s0[g] = 0.f; s1[g] = 0.f; }
    KGroup kcur = load_kgroup(K, kc, 0, ln, lh);
#pragma unroll
    for (int gi = 0; gi < 6; ++gi) {
      KGroup knxt;
      if (gi < 5) knxt = load_kgroup(K, kc, (gi + 1) * 4, ln, lh);
      __builtin_amdgcn_sched_barrier(0);
#pragma unroll
      for (int u = 0; u < 4; ++u) {
        s0 = wmma_bf16(q[gi * 4 + u], kcur.a[u], s0);
        s1 = wmma_bf16(q[gi * 4 + u], kcur.a[4 + u], s1);
      }
      if (gi < 5) kcur = knxt;
    }
    // ---- online softmax (per-row over the 32 keys) ----
    float alpha[8];
#pragma unroll
    for (int g = 0; g < 8; ++g) {
      float mx   = redmax16(fmaxf(s0[g], s1[g]));
      float mnew = fmaxf(ms[g], mx);
      alpha[g]   = exp2f((ms[g] - mnew) * LOG2E);
      float p0   = exp2f((s0[g] - mnew) * LOG2E);
      float p1   = exp2f((s1[g] - mnew) * LOG2E);
      ls[g] = ls[g] * alpha[g] + redsum16(p0 + p1);
      ms[g] = mnew;
      const int m = g + 8 * lh;               // C-layout row for this lane half
      pbuf[wave][m][ln]      = (__bf16)p0;    // P row-major [m][k], k = ln
      pbuf[wave][m][16 + ln] = (__bf16)p1;    // k = 16 + ln
    }
    // reload P as a 16x32 A operand (same-wave LDS round trip, DS in-order)
    const __bf16* pr = &pbuf[wave][ln][lh * 16];
    v16bf P = cat8(*(const v8bf*)pr, *(const v8bf*)(pr + 8));

    // ---- acc = acc*alpha + P . V, pipelined; rescale fills load shadows ----
    VGroup vcur = load_vgroup(Vt, kc, 0, ln, lh);
#pragma unroll
    for (int gi = 0; gi < 12; ++gi) {
      VGroup vnxt;
      if (gi < 11) vnxt = load_vgroup(Vt, kc, (gi + 1) * 4, ln, lh);
      __builtin_amdgcn_sched_barrier(0);
#pragma unroll
      for (int u = 0; u < 4; ++u) {
        const int t = gi * 4 + u;
#pragma unroll
        for (int g = 0; g < 8; ++g) acc[t][g] *= alpha[g];
        acc[t] = wmma_bf16(P, vcur.v[u], acc[t]);
      }
      if (gi < 11) vcur = vnxt;
    }
  }

  // ---- epilogue: 1/l, LayerNorm over d=768, dot with w_out ----
#pragma unroll
  for (int g = 0; g < 8; ++g) {
    const float rl = 1.0f / ls[g];
#pragma unroll
    for (int t = 0; t < 48; ++t) acc[t][g] *= rl;
  }
  float mu[8], rstd[8];
#pragma unroll
  for (int g = 0; g < 8; ++g) {
    float s = 0.f;
#pragma unroll
    for (int t = 0; t < 48; ++t) s += acc[t][g];
    mu[g] = redsum16(s) * (1.0f / 768.0f);
  }
#pragma unroll
  for (int g = 0; g < 8; ++g) {
    float s = 0.f;
#pragma unroll
    for (int t = 0; t < 48; ++t) { float d = acc[t][g] - mu[g]; s += d * d; }
    rstd[g] = rsqrtf(redsum16(s) * (1.0f / 768.0f) + 1e-5f);
  }
  float og[8];
#pragma unroll
  for (int g = 0; g < 8; ++g) og[g] = 0.f;
#pragma unroll
  for (int t = 0; t < 48; ++t) {
    const int d = t * 16 + ln;
    const float ga = gamma[d], be = beta[d], ww = w_out[d];
#pragma unroll
    for (int g = 0; g < 8; ++g)
      og[g] += ((acc[t][g] - mu[g]) * rstd[g] * ga + be) * ww;
  }
  const float bo = b_out[0];
#pragma unroll
  for (int g = 0; g < 8; ++g) {
    const float v = redsum16(og[g]) + bo;
    if (ln == 0) out[row0 + g + 8 * lh] = v;   // out flat index == global row
  }
}

// ---------------------------------------------------------------------------
extern "C" void kernel_launch(void* const* d_in, const int* in_sizes, int n_in,
                              void* d_out, int out_size, void* d_ws, size_t ws_size,
                              hipStream_t stream) {
  (void)in_sizes; (void)n_in; (void)out_size; (void)ws_size;
  const float* x     = (const float*)d_in[0];
  const float* Wq    = (const float*)d_in[1];
  const float* bq    = (const float*)d_in[2];
  const float* Wk    = (const float*)d_in[3];
  const float* bk    = (const float*)d_in[4];
  const float* Wv    = (const float*)d_in[5];
  const float* bv    = (const float*)d_in[6];
  const float* gamma = (const float*)d_in[7];
  const float* beta  = (const float*)d_in[8];
  const float* wout  = (const float*)d_in[9];
  const float* bout  = (const float*)d_in[10];
  uint16_t* ws = (uint16_t*)d_ws;   // needs ~75.5 MB (Qb+Kb+Vtb bf16)

  // 3 mats x 1024 row-tiles x 12 col-groups = 36864 waves, 4 waves/block
  qkv_proj_kernel<<<9216, 128, 0, stream>>>(x, Wq, bq, Wk, bk, Wv, bv, ws);
  // 16384 query rows / (4 waves * 16 rows) = 256 blocks
  flash_attn_ln_kernel<<<256, 128, 0, stream>>>(ws, gamma, beta, wout, bout,
                                                (float*)d_out);
}
